// CBERT_50912542327394
// MI455X (gfx1250) — matrix operand order repaired and checked by reading, not compiled, not run on hardware
//
#include <hip/hip_runtime.h>
#include <hip/hip_bf16.h>
#include <math.h>

// ---------------------------------------------------------------------------
// Types for CDNA5 WMMA (gfx1250, wave32)
// ---------------------------------------------------------------------------
typedef __attribute__((ext_vector_type(8)))  __bf16 v8bf;
typedef __attribute__((ext_vector_type(16))) __bf16 v16bf;
typedef __attribute__((ext_vector_type(8)))  float  v8f;

// f32 -> bf16, round-to-nearest (ties away): add 0x8000 then take high half
__device__ __forceinline__ unsigned short f2bf(float f) {
    unsigned u = __float_as_uint(f) + 0x8000u;
    return (unsigned short)(u >> 16);
}

// packed f32x2 -> bf16x2 in one v_perm_b32:
// result = { high16(b+0x8000), high16(a+0x8000) }
__device__ __forceinline__ unsigned pack2(float a, float b) {
    unsigned ua = __float_as_uint(a) + 0x8000u;
    unsigned ub = __float_as_uint(b) + 0x8000u;
    // V_PERM_B32: sel bytes 0-3 pick from 2nd source, 4-7 from 1st source.
    // D.b0 = ua.b2, D.b1 = ua.b3, D.b2 = ub.b2, D.b3 = ub.b3
    return __builtin_amdgcn_perm(ub, ua, 0x07060302u);
}

// ---------------------------------------------------------------------------
// Generic strided-batched GEMM, fp32 in / fp32 out, bf16 WMMA compute.
//   C[m][n] = act( sum_k A[m][k] * B[k][n] + bias[n] )      (TRB == 0)
//   C[m][n] = act( sum_k A[m][k] * B[n][k] + bias[n] )      (TRB == 1)
// Tile: 128x64 per workgroup, 256 threads = 8 waves, wave does 32x32 (2x2
// WMMA 16x16 tiles). K-tile 64 = two v_wmma_f32_16x16x32_bf16 substeps.
// Double-buffered LDS + register-staged pipeline: one barrier per K-tile.
// Batch: z -> (z/HH, z%HH) with independent strides (covers [b,h] attention
// batching and plain batching).
// Requires: M%128==0, N%64==0, K%64==0 (true for every GEMM in this model).
// ---------------------------------------------------------------------------
template<int TRB>
__global__ __launch_bounds__(256)
void cb_gemm_wmma(const float* __restrict__ A, const float* __restrict__ Bm,
                  const float* __restrict__ bias, float* __restrict__ C,
                  int M, int N, int Kd, int lda, int ldb, int ldc, int act,
                  int HH, long long sA1, long long sA2,
                  long long sB1, long long sB2,
                  long long sC1, long long sC2)
{
    constexpr int BM = 128, BN = 64, KT = 64;
    __shared__ __align__(16) unsigned short As[2][BM * KT];   // [m][k]
    __shared__ __align__(16) unsigned short Bs[2][KT * BN];   // [k][n]

    const long long z  = blockIdx.z;
    const long long z1 = z / HH, z2 = z % HH;
    const float* A_ = A  + z1 * sA1 + z2 * sA2;
    const float* B_ = Bm + z1 * sB1 + z2 * sB2;
    float*       C_ = C  + z1 * sC1 + z2 * sC2;

    const int m0   = blockIdx.y * BM;
    const int n0   = blockIdx.x * BN;
    const int tid  = threadIdx.x;
    const int lane = tid & 31;
    const int wid  = tid >> 5;      // 0..7
    const int wm   = wid >> 1;      // 0..3  -> 32-row slab
    const int wn   = wid & 1;       // 0..1  -> 32-col slab
    const int lr   = lane & 15;
    const int kA   = (lane >> 4) * 8;   // A-frag K chunk start per ISA layout

    // staging assignments
    const int ar = tid >> 1, ah = tid & 1;   // A: 128 rows x 2 k-halves of 32
    const float* aRow = A_ + (long long)(m0 + ar) * lda + ah * 32;
    const int br = tid >> 2, bq = tid & 3;   // B: 64 rows x 4 chunks of 16
    const float* bRow = (TRB == 0)
        ? B_ + (long long)br * ldb + n0 + bq * 16    // row k=br, n chunk
        : B_ + (long long)(n0 + br) * ldb + bq * 16; // row n=br, k chunk

    float4 rA[8], rB[4];

    auto loadStage = [&](int kb) {
        #pragma unroll
        for (int j = 0; j < 8; ++j)
            rA[j] = *(const float4*)(aRow + kb + j * 4);
        if (TRB == 0) {
            #pragma unroll
            for (int j = 0; j < 4; ++j)
                rB[j] = *(const float4*)(bRow + (long long)kb * ldb + j * 4);
        } else {
            #pragma unroll
            for (int j = 0; j < 4; ++j)
                rB[j] = *(const float4*)(bRow + kb + j * 4);
        }
    };

    auto storeStage = [&](int buf) {
        unsigned* da = (unsigned*)(As[buf] + ar * KT + ah * 32);
        #pragma unroll
        for (int j = 0; j < 8; ++j) {
            da[j * 2 + 0] = pack2(rA[j].x, rA[j].y);
            da[j * 2 + 1] = pack2(rA[j].z, rA[j].w);
        }
        if (TRB == 0) {
            unsigned* db = (unsigned*)(Bs[buf] + br * BN + bq * 16);
            #pragma unroll
            for (int j = 0; j < 4; ++j) {
                db[j * 2 + 0] = pack2(rB[j].x, rB[j].y);
                db[j * 2 + 1] = pack2(rB[j].z, rB[j].w);
            }
        } else {   // transpose 16 k-values into [k][n]
            #pragma unroll
            for (int j = 0; j < 4; ++j) {
                float fv[4] = {rB[j].x, rB[j].y, rB[j].z, rB[j].w};
                #pragma unroll
                for (int u = 0; u < 4; ++u)
                    Bs[buf][(bq * 16 + j * 4 + u) * BN + br] = f2bf(fv[u]);
            }
        }
    };

    v8f acc[2][2];
    const v8f vz = {0.f,0.f,0.f,0.f,0.f,0.f,0.f,0.f};
    acc[0][0] = vz; acc[0][1] = vz; acc[1][0] = vz; acc[1][1] = vz;

    auto compute = [&](int buf) {
        #pragma unroll
        for (int ks = 0; ks < 2; ++ks) {   // two 32-deep WMMA substeps
            v16bf afr[2], bfr[2];
            #pragma unroll
            for (int mi = 0; mi < 2; ++mi) {
                const v8bf* ap = (const v8bf*)(As[buf] +
                    (wm * 32 + mi * 16 + lr) * KT + ks * 32 + kA);
                v8bf lo = ap[0], hi = ap[2];   // K chunk, K chunk + 16
                afr[mi] = __builtin_shufflevector(lo, hi,
                    0,1,2,3,4,5,6,7,8,9,10,11,12,13,14,15);
            }
            #pragma unroll
            for (int ni = 0; ni < 2; ++ni) {
                const unsigned short* bp = Bs[buf] +
                    (ks * 32 + kA) * BN + wn * 32 + ni * 16 + lr;
                #pragma unroll
                for (int s = 0; s < 16; ++s)
                    bfr[ni][s] = __builtin_bit_cast(__bf16, bp[(s + (s >> 3) * 8) * BN]);
            }
            #pragma unroll
            for (int mi = 0; mi < 2; ++mi)
                #pragma unroll
                for (int ni = 0; ni < 2; ++ni)
                    acc[mi][ni] = __builtin_amdgcn_wmma_f32_16x16x32_bf16(
                        false, afr[mi], false, bfr[ni], (short)0,
                        acc[mi][ni], false, false);
        }
    };

    // software pipeline: global loads for stage k+1 overlap WMMA on stage k
    loadStage(0);
    storeStage(0);
    int buf = 0;
    for (int kb = 0; kb < Kd; kb += KT) {
        const bool notlast = (kb + KT < Kd);
        if (notlast) loadStage(kb + KT);
        __syncthreads();                 // stage 'buf' visible to all waves
        compute(buf);
        if (notlast) storeStage(buf ^ 1);
        buf ^= 1;
    }

    // epilogue: C/D layout = 8 VGPRs, lanes 0-15 M=r, lanes 16-31 M=8+r
    #pragma unroll
    for (int mi = 0; mi < 2; ++mi) {
        #pragma unroll
        for (int ni = 0; ni < 2; ++ni) {
            int col = n0 + wn * 32 + ni * 16 + lr;
            float bv = bias ? bias[col] : 0.f;
            #pragma unroll
            for (int r = 0; r < 8; ++r) {
                long long row = m0 + wm * 32 + mi * 16 + (lane >> 4) * 8 + r;
                float v = acc[mi][ni][r] + bv;
                if (act == 1)       v = 0.5f * v * (1.0f + erff(v * 0.70710678118f)); // exact gelu
                else if (act == 2)  v = fmaxf(v, 0.f);
                C_[row * ldc + col] = v;
            }
        }
    }
}

// ---------------------------------------------------------------------------
// Small support kernels
// ---------------------------------------------------------------------------
__global__ void cb_zero(float* p, long long n) {
    long long i = (long long)blockIdx.x * blockDim.x + threadIdx.x;
    if (i < n) p[i] = 0.f;
}

__global__ void cb_embed(float* hid, const int* ids, const float* we,
                         const float* pe, int L, int D) {
    int bl = blockIdx.x;
    int l  = bl % L;
    int id = (l < 2) ? 0 : ids[bl];
    const float* w = we + (long long)id * D;
    for (int d = threadIdx.x; d < D; d += blockDim.x)
        hid[(long long)bl * D + d] = (l < 2) ? w[d] : (w[d] + pe[l * D + d]);
}

__global__ void cb_copy_cls(float* dst, const float* hid, int L, int D, int B) {
    int i = blockIdx.x * blockDim.x + threadIdx.x;
    if (i >= B * D) return;
    int b = i / D, d = i % D;
    dst[i] = hid[(long long)b * L * D + d];
}

__global__ void cb_qcross(float* q, const float* ccls, const float* Wq,
                          const float* bq, int B, int L, int D) {
    int i = blockIdx.x * blockDim.x + threadIdx.x;
    if (i >= B * D) return;
    int b = i / D, d = i % D;
    float a = bq[d];
    const float* c = ccls + (long long)b * D;
    for (int k = 0; k < D; ++k) a += c[k] * Wq[(long long)k * D + d];
    q[((long long)b * L + 1) * D + d] = a;
}

// row = b*H*L + h*L + qpos ; L == 512, block 256
__global__ void cb_softmax(float* S, const int* mask, int H, int L, float scale) {
    long long row = blockIdx.x;
    int b = (int)(row / ((long long)H * L));
    float* sr = S + row * (long long)L;
    const int* mr = mask + (long long)b * L;
    int tid = threadIdx.x;
    __shared__ float red[256];
    float lv[2]; float mx = -1e30f;
    #pragma unroll
    for (int j = 0; j < 2; ++j) {
        int k = tid + j * 256;
        float t = sr[k] * scale + (1.0f - (float)mr[k]) * -10000.0f;
        lv[j] = t; mx = fmaxf(mx, t);
    }
    red[tid] = mx; __syncthreads();
    for (int off = 128; off; off >>= 1) {
        if (tid < off) red[tid] = fmaxf(red[tid], red[tid + off]);
        __syncthreads();
    }
    mx = red[0]; __syncthreads();
    float s = 0.f;
    #pragma unroll
    for (int j = 0; j < 2; ++j) { lv[j] = __expf(lv[j] - mx); s += lv[j]; }
    red[tid] = s; __syncthreads();
    for (int off = 128; off; off >>= 1) {
        if (tid < off) red[tid] += red[tid + off];
        __syncthreads();
    }
    float inv = 1.0f / red[0];
    #pragma unroll
    for (int j = 0; j < 2; ++j) sr[tid + j * 256] = lv[j] * inv;
}

// x = LN(x + y) * g + b ; D == 768, block 256
__global__ void cb_add_ln(float* x, const float* y, const float* g,
                          const float* b, int D) {
    long long row = blockIdx.x;
    float* xr = x + row * D;
    const float* yr = y + row * D;
    int tid = threadIdx.x;
    __shared__ float r1[256], r2[256];
    float v[3]; float s = 0.f, s2 = 0.f;
    #pragma unroll
    for (int j = 0; j < 3; ++j) {
        int d = tid + j * 256;
        float t = xr[d] + yr[d];
        v[j] = t; s += t; s2 += t * t;
    }
    r1[tid] = s; r2[tid] = s2; __syncthreads();
    for (int off = 128; off; off >>= 1) {
        if (tid < off) { r1[tid] += r1[tid + off]; r2[tid] += r2[tid + off]; }
        __syncthreads();
    }
    float mean = r1[0] / D;
    float var  = r2[0] / D - mean * mean;
    float rs   = rsqrtf(var + 1e-12f);
    #pragma unroll
    for (int j = 0; j < 3; ++j) {
        int d = tid + j * 256;
        xr[d] = (v[j] - mean) * rs * g[d] + b[d];
    }
}

__global__ void cb_gate(float* gl, const float* x, const float* gate,
                        int T, int D, int E) {
    int i = blockIdx.x * blockDim.x + threadIdx.x;
    if (i >= T * E) return;
    int t = i / E, e = i % E;
    const float* xr = x + (long long)t * D;
    float a = 0.f;
    for (int k = 0; k < D; ++k) a += xr[k] * gate[k * E + e];
    gl[i] = a;
}

// softmax over E=8, top-2, combine weights; probs back into gl; top0 as float
__global__ void cb_gate_fin(float* gl, float* comb, float* top0, int T, int E) {
    int t = blockIdx.x * blockDim.x + threadIdx.x;
    if (t >= T) return;
    float p[8]; float mx = -1e30f;
    for (int e = 0; e < 8; ++e) { p[e] = gl[t * 8 + e]; mx = fmaxf(mx, p[e]); }
    float s = 0.f;
    for (int e = 0; e < 8; ++e) { p[e] = __expf(p[e] - mx); s += p[e]; }
    float inv = 1.f / s;
    for (int e = 0; e < 8; ++e) { p[e] *= inv; gl[t * 8 + e] = p[e]; comb[t * 8 + e] = 0.f; }
    int i0 = 0;
    for (int e = 1; e < 8; ++e) if (p[e] > p[i0]) i0 = e;
    int i1 = (i0 == 0) ? 1 : 0;
    for (int e = 0; e < 8; ++e) if (e != i0 && p[e] > p[i1]) i1 = e;
    float w = p[i0] + p[i1];
    comb[t * 8 + i0] = p[i0] / w;
    comb[t * 8 + i1] = p[i1] / w;
    top0[t] = (float)i0;
}

// deterministic aux: total += E * sum_e mean(probs[:,e]) * mean(top0==e)
__global__ void cb_aux(const float* probs, const float* top0, float* total,
                       int T, int E) {
    __shared__ float r1[256], r2[256];
    int tid = threadIdx.x;
    float acc = 0.f;
    for (int e = 0; e < E; ++e) {
        float a = 0.f, c = 0.f;
        for (int t = tid; t < T; t += 256) {
            a += probs[t * E + e];
            c += (top0[t] == (float)e) ? 1.f : 0.f;
        }
        r1[tid] = a; r2[tid] = c; __syncthreads();
        for (int off = 128; off; off >>= 1) {
            if (tid < off) { r1[tid] += r1[tid + off]; r2[tid] += r2[tid + off]; }
            __syncthreads();
        }
        if (tid == 0) acc += (r1[0] / T) * (r2[0] / T);
        __syncthreads();
    }
    if (tid == 0) total[0] += (float)E * acc;
}

__global__ void cb_moe_accum(float* outp, const float* y, const float* comb,
                             int e, int E, int D, long long n) {
    long long i = (long long)blockIdx.x * blockDim.x + threadIdx.x;
    if (i >= n) return;
    long long t = i / D;
    outp[i] += comb[t * E + e] * y[i];
}

__global__ void cb_pool(float* out, const float* hid, const int* mask,
                        int B, int L, int D) {
    int i = blockIdx.x * blockDim.x + threadIdx.x;
    if (i >= B * D) return;
    int b = i / D, d = i % D;
    float s = 0.f, ms = 0.f;
    for (int l = 0; l < L; ++l) {
        float m = (float)mask[b * L + l];
        s  += hid[((long long)b * L + l) * D + d] * m;
        ms += m;
    }
    out[i] = s / fmaxf(ms, 1e-6f);
}

__global__ void cb_tag(float* out16, const float* tags, const float* W1,
                       const float* b1, const float* W2, const float* b2) {
    int b = blockIdx.x, t = threadIdx.x;
    __shared__ float h[32];
    float a = b1[t];
    for (int k = 0; k < 32; ++k) a += tags[b * 32 + k] * W1[k * 32 + t];
    h[t] = fmaxf(a, 0.f);
    __syncthreads();
    if (t < 16) {
        float o = b2[t];
        for (int k = 0; k < 32; ++k) o += h[k] * W2[k * 16 + t];
        out16[b * 16 + t] = o;
    }
}

__global__ void cb_concat(float* z, const float* tv, const float* cv,
                          const float* feats, const float* tg,
                          int B, int D, int CIN) {
    int i = blockIdx.x * blockDim.x + threadIdx.x;
    if (i >= B * CIN) return;
    int b = i / CIN, j = i % CIN;
    float v;
    if (j < D)              v = tv[b * D + j];
    else if (j < 2 * D)     v = cv[b * D + (j - D)];
    else if (j < 2 * D + 3) v = feats[b * 3 + (j - 2 * D)];
    else                    v = tg[b * 16 + (j - 2 * D - 3)];
    z[i] = v;
}

__global__ void cb_mlp(float* out, const float* z, const float* W,
                       const float* bias, int B, int Kd, int N, int relu) {
    int i = blockIdx.x * blockDim.x + threadIdx.x;
    if (i >= B * N) return;
    int b = i / N, n = i % N;
    float a = bias[n];
    const float* zr = z + (long long)b * Kd;
    for (int k = 0; k < Kd; ++k) a += zr[k] * W[(long long)k * N + n];
    if (relu) a = fmaxf(a, 0.f);
    out[i] = a;
}

// ---------------------------------------------------------------------------
// Host-side launch helper
// ---------------------------------------------------------------------------
static inline void launch_gemm(hipStream_t st, int trb,
    const float* A, const float* B, const float* bias, float* C,
    int M, int N, int Kd, int lda, int ldb, int ldc, int act,
    int nz = 1, int HH = 1,
    long long sA1 = 0, long long sA2 = 0,
    long long sB1 = 0, long long sB2 = 0,
    long long sC1 = 0, long long sC2 = 0)
{
    dim3 g(N / 64, M / 128, nz), blk(256);
    if (trb)
        cb_gemm_wmma<1><<<g, blk, 0, st>>>(A, B, bias, C, M, N, Kd, lda, ldb, ldc,
                                           act, HH, sA1, sA2, sB1, sB2, sC1, sC2);
    else
        cb_gemm_wmma<0><<<g, blk, 0, st>>>(A, B, bias, C, M, N, Kd, lda, ldb, ldc,
                                           act, HH, sA1, sA2, sB1, sB2, sC1, sC2);
}

extern "C" void kernel_launch(void* const* d_in, const int* in_sizes, int n_in,
                              void* d_out, int out_size, void* d_ws, size_t ws_size,
                              hipStream_t stream)
{
    (void)in_sizes; (void)n_in; (void)out_size; (void)ws_size;
    const int B = 4, L = 512, D = 768, H = 12, HD = 64, F = 3072, E = 8, NL = 2;
    const int T = B * L;
    const int CIN = 2 * D + 3 + 16;   // 1555

    auto F32 = [&](int i) { return (const float*)d_in[i]; };
    auto I32 = [&](int i) { return (const int*)d_in[i]; };

    const int* text_ids  = I32(0);
    const int* text_mask = I32(1);
    const int* code_ids  = I32(2);
    const int* code_mask = I32(3);
    const float* feats   = F32(4);
    const float* tags    = F32(5);
    // tower leaves in dict insertion order: we,pe,Wq,bq,Wk,bk,Wv,bv,Wo,bo,
    // ln1_g,ln1_b,gate,fc1,b1,fc2,b2,ln2_g,ln2_b  (19 leaves)
    const int TP = 6, CP = 25, HP = 44;

    float* out = (float*)d_out;       // [0..3] logits, [4] aux total

    // workspace carve-out (fp32 elements)
    float* p = (float*)d_ws;
    auto alloc = [&](size_t n) { float* r = p; p += n; return r; };
    float* t_hid  = alloc((size_t)T * D);
    float* c_hid  = alloc((size_t)T * D);
    float* dq     = alloc((size_t)T * D);
    float* dk     = alloc((size_t)T * D);
    float* dv     = alloc((size_t)T * D);
    float* dctx   = alloc((size_t)T * D);
    float* dob    = alloc((size_t)T * D);
    float* dscore = alloc((size_t)B * H * L * L);
    float* dh     = alloc((size_t)T * F);
    float* dy     = alloc((size_t)T * D);
    float* df     = alloc((size_t)T * D);
    float* clsA   = alloc((size_t)B * D);
    float* clsB   = alloc((size_t)B * D);
    float* gl     = alloc((size_t)T * E);
    float* comb   = alloc((size_t)T * E);
    float* top0   = alloc((size_t)T);
    float* tv     = alloc((size_t)B * D);
    float* cv     = alloc((size_t)B * D);
    float* tg     = alloc((size_t)B * 16);
    float* zb     = alloc((size_t)B * CIN);
    float* h1     = alloc((size_t)B * 2 * D);
    float* h2     = alloc((size_t)B * D);

    cb_zero<<<1, 32, 0, stream>>>(out + 4, 1);

    cb_embed<<<T, 256, 0, stream>>>(t_hid, text_ids, F32(TP + 0), F32(TP + 1), L, D);
    cb_embed<<<T, 256, 0, stream>>>(c_hid, code_ids, F32(CP + 0), F32(CP + 1), L, D);

    auto run_layer = [&](float* hid, const int* mask, const float* ccls, int PB, int i) {
        const float* Wq = F32(PB + 2)  + (size_t)i * D * D;
        const float* bq = F32(PB + 3)  + (size_t)i * D;
        const float* Wk = F32(PB + 4)  + (size_t)i * D * D;
        const float* bk = F32(PB + 5)  + (size_t)i * D;
        const float* Wv = F32(PB + 6)  + (size_t)i * D * D;
        const float* bv = F32(PB + 7)  + (size_t)i * D;
        const float* Wo = F32(PB + 8)  + (size_t)i * D * D;
        const float* bo = F32(PB + 9)  + (size_t)i * D;

        launch_gemm(stream, 0, hid, Wq, bq, dq, T, D, D, D, D, D, 0);
        cb_qcross<<<(B * D + 255) / 256, 256, 0, stream>>>(dq, ccls, Wq, bq, B, L, D);
        launch_gemm(stream, 0, hid, Wk, bk, dk, T, D, D, D, D, D, 0);
        launch_gemm(stream, 0, hid, Wv, bv, dv, T, D, D, D, D, D, 0);

        // scores[b,h] = Q[b,:,h*HD:] @ K[b,:,h*HD:]^T   (48 batched NT GEMMs)
        launch_gemm(stream, 1, dq, dk, nullptr, dscore, L, L, HD, D, D, L, 0,
                    B * H, H,
                    (long long)L * D, HD,
                    (long long)L * D, HD,
                    (long long)H * L * L, (long long)L * L);
        cb_softmax<<<B * H * L, 256, 0, stream>>>(dscore, mask, H, L, 0.125f);
        // ctx[b,h] = P[b,h] @ V[b,:,h*HD:]
        launch_gemm(stream, 0, dscore, dv, nullptr, dctx, L, HD, L, L, D, D, 0,
                    B * H, H,
                    (long long)H * L * L, (long long)L * L,
                    (long long)L * D, HD,
                    (long long)L * D, HD);
        launch_gemm(stream, 0, dctx, Wo, bo, dob, T, D, D, D, D, D, 0);
        cb_add_ln<<<T, 256, 0, stream>>>(hid, dob, F32(PB + 10) + (size_t)i * D,
                                         F32(PB + 11) + (size_t)i * D, D);

        // MoE gating
        cb_gate<<<(T * E + 255) / 256, 256, 0, stream>>>(gl, hid,
            F32(PB + 12) + (size_t)i * D * E, T, D, E);
        cb_gate_fin<<<(T + 255) / 256, 256, 0, stream>>>(gl, comb, top0, T, E);
        cb_aux<<<1, 256, 0, stream>>>(gl, top0, out + 4, T, E);

        cb_zero<<<((long long)T * D + 255) / 256, 256, 0, stream>>>(df, (long long)T * D);
        for (int e = 0; e < E; ++e) {
            const float* fc1 = F32(PB + 13) + (size_t)(i * E + e) * D * F;
            const float* b1  = F32(PB + 14) + (size_t)(i * E + e) * F;
            const float* fc2 = F32(PB + 15) + (size_t)(i * E + e) * F * D;
            const float* b2  = F32(PB + 16) + (size_t)(i * E + e) * D;
            launch_gemm(stream, 0, hid, fc1, b1, dh, T, F, D, D, F, F, 1); // gelu
            launch_gemm(stream, 0, dh, fc2, b2, dy, T, D, F, F, D, D, 0);
            cb_moe_accum<<<((long long)T * D + 255) / 256, 256, 0, stream>>>(
                df, dy, comb, e, E, D, (long long)T * D);
        }
        cb_add_ln<<<T, 256, 0, stream>>>(hid, df, F32(PB + 17) + (size_t)i * D,
                                         F32(PB + 18) + (size_t)i * D, D);
    };

    for (int i = 0; i < NL; ++i) {
        cb_copy_cls<<<(B * D + 255) / 256, 256, 0, stream>>>(clsA, t_hid, L, D, B);
        cb_copy_cls<<<(B * D + 255) / 256, 256, 0, stream>>>(clsB, c_hid, L, D, B);
        run_layer(t_hid, text_mask, clsB, TP, i);
        run_layer(c_hid, code_mask, clsA, CP, i);
    }

    cb_pool<<<(B * D + 255) / 256, 256, 0, stream>>>(tv, t_hid, text_mask, B, L, D);
    cb_pool<<<(B * D + 255) / 256, 256, 0, stream>>>(cv, c_hid, code_mask, B, L, D);
    cb_tag<<<B, 32, 0, stream>>>(tg, tags, F32(HP + 0), F32(HP + 1), F32(HP + 2), F32(HP + 3));
    cb_concat<<<(B * CIN + 255) / 256, 256, 0, stream>>>(zb, tv, cv, feats, tg, B, D, CIN);
    cb_mlp<<<(B * 2 * D + 255) / 256, 256, 0, stream>>>(h1, zb, F32(HP + 4), F32(HP + 5), B, CIN, 2 * D, 1);
    cb_mlp<<<(B * D + 255) / 256, 256, 0, stream>>>(h2, h1, F32(HP + 6), F32(HP + 7), B, 2 * D, D, 1);
    cb_mlp<<<1, 32, 0, stream>>>(out, h2, F32(HP + 8), F32(HP + 9), B, D, 1, 0);
}
